// DynamicGraphIpaFrameDenoiser_8323646620403
// MI455X (gfx1250) — matrix operand order repaired and checked by reading, not compiled, not run on hardware
//
#include <hip/hip_runtime.h>
#include <hip/hip_bf16.h>

// ---------------- problem constants (must match reference) ----------------
#define NN   16384
#define EE   393216
#define ESN  131072
#define CSD  256
#define CLD  128
#define CZD  128
#define EIND 192
#define HH   8
#define CHD  32
#define PQD  8
#define PVD  12

typedef __attribute__((ext_vector_type(16))) __bf16 v16bf;
typedef __attribute__((ext_vector_type(8)))  float  v8f;

// ---------------- workspace layout (floats; every region multiple of 64 floats) ----------------
static constexpr size_t OF_Z   = 0;                           // z [E,128]
static constexpr size_t OF_H   = OF_Z   + (size_t)EE*CZD;     // MLP scratch / abuf [E,8] / cat [N,1664]
static constexpr size_t OF_Q   = OF_H   + (size_t)EE*CZD;
static constexpr size_t OF_K   = OF_Q   + (size_t)NN*256;
static constexpr size_t OF_V   = OF_K   + (size_t)NN*256;
static constexpr size_t OF_QP  = OF_V   + (size_t)NN*256;
static constexpr size_t OF_KP  = OF_QP  + (size_t)NN*192;
static constexpr size_t OF_VP  = OF_KP  + (size_t)NN*192;
static constexpr size_t OF_QPG = OF_VP  + (size_t)NN*288;
static constexpr size_t OF_KPG = OF_QPG + (size_t)NN*192;
static constexpr size_t OF_VPG = OF_KPG + (size_t)NN*192;
static constexpr size_t OF_MX  = OF_VPG + (size_t)NN*288;     // N*H (uint then float)
static constexpr size_t OF_DEN = OF_MX  + (size_t)NN*HH;
static constexpr size_t OF_OAC = OF_DEN + (size_t)NN*HH;      // [N,256]
static constexpr size_t OF_OPG = OF_OAC + (size_t)NN*256;     // [N,288]
static constexpr size_t OF_OPR = OF_OPG + (size_t)NN*288;     // [N,1024]
static constexpr size_t OF_NF  = OF_OPR + (size_t)NN*1024;
static constexpr size_t OF_UPD = OF_NF  + (size_t)NN*256;
static constexpr size_t OF_T1  = OF_UPD + (size_t)NN*256;     // also concat [N,384] (with T2)
static constexpr size_t OF_T2  = OF_T1  + (size_t)NN*256;
static constexpr size_t OF_U6  = OF_T2  + (size_t)NN*256;
static constexpr size_t OF_BA  = OF_U6  + (size_t)NN*64;      // bf16 A staging: EE*192 ushorts
static constexpr size_t OF_BW  = OF_BA  + (size_t)EE*96;      // bf16 W^T staging: 1664*256 ushorts
static constexpr size_t WS_FLOATS = OF_BW + (size_t)212992;

// ---------------- device helpers ----------------
__device__ __forceinline__ unsigned short f2bf(float f) {
  unsigned u = __float_as_uint(f);
  unsigned r = u + 0x7FFFu + ((u >> 16) & 1u);          // round-to-nearest-even
  return (unsigned short)(r >> 16);
}
__device__ __forceinline__ unsigned fenc(float f) {      // monotonic float->uint
  unsigned u = __float_as_uint(f);
  return (u & 0x80000000u) ? ~u : (u | 0x80000000u);
}

// ---------------- precision staging kernels (run once per GEMM, bandwidth bound) ----------------
__global__ void cvtbf_kernel(const float* __restrict__ s, unsigned short* __restrict__ d, long n) {
  long i = (long)blockIdx.x * blockDim.x + threadIdx.x;
  if (i < n) d[i] = f2bf(s[i]);
}
// W [K,N] fp32 row-major -> Wt [N,K] bf16
__global__ void cvttr_kernel(const float* __restrict__ W, unsigned short* __restrict__ Wt,
                             int K, int N) {
  long i = (long)blockIdx.x * blockDim.x + threadIdx.x;
  if (i >= (long)K * N) return;
  int k = (int)(i / N), n = (int)(i % N);
  Wt[(size_t)n * K + k] = f2bf(W[i]);
}

// ---------------- bf16 WMMA GEMM: C = act(A[M,K] @ W[K,N] + bias [+ resid]) ----------------
// A: bf16 row-major, Wt: bf16 [N,K]. One wave computes a 16x32 tile; K unrolled by 64.
// Fragments map to the documented gfx1250 WMMA VGPR layouts:
//  A 16x32: lanes 0-15 K 0-7 / 16-23, lanes 16-31 K 8-15 / 24-31  -> two contiguous b128 loads
//  B 32x16: lanes 0-15 K 0-7 / 16-23, lanes 16-31 K 8-15 / 24-31 (col = lane&15) -> two contiguous b128 loads
union Frag { v16bf v; uint4 q[2]; };
union CFr  { v8f v; float f[8]; };

__global__ void gemm_bf16_kernel(const unsigned short* __restrict__ A,
                                 const unsigned short* __restrict__ Wt,
                                 const float* __restrict__ bias, const float* __restrict__ resid,
                                 float* __restrict__ C, int M, int K, int N,
                                 int tilesN2, int tilesTotal, int act)
{
  int tile = blockIdx.x * (blockDim.x >> 5) + (threadIdx.x >> 5);
  if (tile >= tilesTotal) return;
  int lane = threadIdx.x & 31;
  int half = lane >> 4;
  int lm   = lane & 15;
  int tm = tile / tilesN2, tn = tile % tilesN2;

  const unsigned short* Arow = A + (size_t)(tm * 16 + lm) * K;
  int col0 = tn * 32 + lm, col1 = col0 + 16;
  const unsigned short* W0 = Wt + (size_t)col0 * K;
  const unsigned short* W1 = Wt + (size_t)col1 * K;

  int aoff0 = half * 8, aoff1 = 16 + half * 8;   // A fragment K offsets
  int boff  = half * 8;                          // B fragment K offset

  Frag a0, a1, b00, b01, b10, b11;
  CFr c0, c1;
#pragma unroll
  for (int r = 0; r < 8; ++r) { c0.f[r] = 0.f; c1.f[r] = 0.f; }

  for (int k0 = 0; k0 < K; k0 += 64) {
    if (k0 + 64 < K) __builtin_prefetch(Arow + k0 + 64, 0, 0);
    a0.q[0]  = *(const uint4*)(Arow + k0 + aoff0);
    a0.q[1]  = *(const uint4*)(Arow + k0 + aoff1);
    a1.q[0]  = *(const uint4*)(Arow + k0 + 32 + aoff0);
    a1.q[1]  = *(const uint4*)(Arow + k0 + 32 + aoff1);
    b00.q[0] = *(const uint4*)(W0 + k0 + boff);
    b00.q[1] = *(const uint4*)(W0 + k0 + boff + 16);
    b01.q[0] = *(const uint4*)(W0 + k0 + 32 + boff);
    b01.q[1] = *(const uint4*)(W0 + k0 + 32 + boff + 16);
    b10.q[0] = *(const uint4*)(W1 + k0 + boff);
    b10.q[1] = *(const uint4*)(W1 + k0 + boff + 16);
    b11.q[0] = *(const uint4*)(W1 + k0 + 32 + boff);
    b11.q[1] = *(const uint4*)(W1 + k0 + 32 + boff + 16);
    c0.v = __builtin_amdgcn_wmma_f32_16x16x32_bf16(false, a0.v, false, b00.v, (short)0, c0.v, false, false);
    c1.v = __builtin_amdgcn_wmma_f32_16x16x32_bf16(false, a0.v, false, b10.v, (short)0, c1.v, false, false);
    c0.v = __builtin_amdgcn_wmma_f32_16x16x32_bf16(false, a1.v, false, b01.v, (short)0, c0.v, false, false);
    c1.v = __builtin_amdgcn_wmma_f32_16x16x32_bf16(false, a1.v, false, b11.v, (short)0, c1.v, false, false);
  }

  float bv0 = bias ? bias[col0] : 0.f;
  float bv1 = bias ? bias[col1] : 0.f;
#pragma unroll
  for (int r = 0; r < 8; ++r) {
    int orow = tm * 16 + r + half * 8;   // C/D layout: vgpr r -> M=r (lanes<16) / r+8
    float v0 = c0.f[r] + bv0;
    float v1 = c1.f[r] + bv1;
    if (resid) {
      v0 += resid[(size_t)orow * N + col0];
      v1 += resid[(size_t)orow * N + col1];
    }
    if (act) { v0 = fmaxf(v0, 0.f); v1 = fmaxf(v1, 0.f); }
    C[(size_t)orow * N + col0] = v0;
    C[(size_t)orow * N + col1] = v1;
  }
}

// ---------------- layernorm (wave per row), optional residual, in-place safe ----------------
__global__ void ln_kernel(const float* __restrict__ x, const float* __restrict__ resid,
                          const float* __restrict__ g, const float* __restrict__ b,
                          float* __restrict__ out, int M, int D)
{
  int row = blockIdx.x * (blockDim.x >> 5) + (threadIdx.x >> 5);
  if (row >= M) return;
  int lane = threadIdx.x & 31;
  const float* xr = x + (size_t)row * D;
  const float* rr = resid ? resid + (size_t)row * D : nullptr;
  float s = 0.f, ss = 0.f;
  for (int d = lane; d < D; d += 32) {
    float v = xr[d] + (rr ? rr[d] : 0.f);
    s += v; ss += v * v;
  }
#pragma unroll
  for (int o = 16; o > 0; o >>= 1) { s += __shfl_xor(s, o, 32); ss += __shfl_xor(ss, o, 32); }
  float m = s / (float)D;
  float var = fmaxf(ss / (float)D - m * m, 0.f);
  float inv = rsqrtf(var + 1e-5f);
  float* orow = out + (size_t)row * D;
  for (int d = lane; d < D; d += 32) {
    float v = xr[d] + (rr ? rr[d] : 0.f);
    orow[d] = (v - m) * inv * g[d] + b[d];
  }
}

// ---------------- small elementwise kernels ----------------
__global__ void fillf_kernel(float* p, float v, long n) {
  long i = (long)blockIdx.x * blockDim.x + threadIdx.x; if (i < n) p[i] = v;
}
__global__ void fillu_kernel(unsigned* p, unsigned v, long n) {
  long i = (long)blockIdx.x * blockDim.x + threadIdx.x; if (i < n) p[i] = v;
}
__global__ void concat_kernel(const float* __restrict__ a, const float* __restrict__ b,
                              float* __restrict__ o, int M, int Da, int Db) {
  long i = (long)blockIdx.x * blockDim.x + threadIdx.x;
  int Dt = Da + Db;
  if (i >= (long)M * Dt) return;
  int r = (int)(i / Dt), c = (int)(i % Dt);
  o[i] = (c < Da) ? a[(size_t)r * Da + c] : b[(size_t)r * Db + (c - Da)];
}
__global__ void scale_rows_kernel(const float* __restrict__ src, const float* __restrict__ s,
                                  float* __restrict__ dst, int M, int D) {
  long i = (long)blockIdx.x * blockDim.x + threadIdx.x;
  if (i >= (long)M * D) return;
  dst[i] = src[i] * s[i / D];
}
__global__ void copy_cols_kernel(const float* __restrict__ src, float* __restrict__ dst,
                                 int M, int Ds, int stride, int off) {
  long i = (long)blockIdx.x * blockDim.x + threadIdx.x;
  if (i >= (long)M * Ds) return;
  int r = (int)(i / Ds), c = (int)(i % Ds);
  dst[(size_t)r * stride + off + c] = src[i];
}

// points [M,P,3] local -> global frame: out = R * p + t
__global__ void frame_apply_kernel(const float* __restrict__ pts, const float* __restrict__ rot,
                                   const float* __restrict__ trans, float* __restrict__ out,
                                   int M, int P) {
  long i = (long)blockIdx.x * blockDim.x + threadIdx.x;
  if (i >= (long)M * P) return;
  int n = (int)(i / P);
  const float* q = pts + i * 3;
  const float* R = rot + (size_t)n * 9;
  const float* t = trans + (size_t)n * 3;
  float x = q[0], y = q[1], z = q[2];
  out[i * 3 + 0] = R[0] * x + R[1] * y + R[2] * z + t[0];
  out[i * 3 + 1] = R[3] * x + R[4] * y + R[5] * z + t[1];
  out[i * 3 + 2] = R[6] * x + R[7] * y + R[8] * z + t[2];
}

// attention logits per (edge, head) + segment max (monotonic-uint atomicMax)
__global__ void logits_kernel(const float* __restrict__ q, const float* __restrict__ k,
                              const float* __restrict__ qpg, const float* __restrict__ kpg,
                              const float* __restrict__ z, const float* __restrict__ bzW,
                              const float* __restrict__ bzb, const float* __restrict__ hwp,
                              const float* __restrict__ rm, const int* __restrict__ ei,
                              float* __restrict__ abuf, unsigned* __restrict__ mx, int Ecnt) {
  long i = (long)blockIdx.x * blockDim.x + threadIdx.x;
  if (i >= (long)Ecnt * HH) return;
  int e = (int)(i / HH), h = (int)(i % HH);
  int row = ei[e], col = ei[Ecnt + e];
  const float* qh = q + (size_t)row * 256 + h * 32;
  const float* kh = k + (size_t)col * 256 + h * 32;
  float dot = 0.f;
#pragma unroll
  for (int c = 0; c < 32; ++c) dot += qh[c] * kh[c];
  float a = dot * 0.10206207262f;                        // sqrt(1/(3*CH))
  float bz = bzb[h];
  const float* ze = z + (size_t)e * CZD;
  for (int c = 0; c < CZD; ++c) bz += ze[c] * bzW[c * HH + h];
  a += 0.57735026919f * bz;                              // sqrt(1/3)
  const float* qp = qpg + (size_t)row * 192 + h * 24;
  const float* kp = kpg + (size_t)col * 192 + h * 24;
  float d2 = 0.f;
#pragma unroll
  for (int j = 0; j < 24; ++j) { float d = qp[j] - kp[j]; d2 += d * d; }
  float hw = log1pf(expf(hwp[h]));                       // softplus
  a += d2 * hw * (-0.04811252243f);                      // -0.5*sqrt(1/(3*PQ*4.5))
  a += (rm[col] - 1.f) * 1e5f;
  abuf[i] = a;
  atomicMax(mx + (size_t)row * HH + h, fenc(a));
}

// decode segment-max in place (uint encoded -> float bits), non-finite -> 0
__global__ void mxfix_kernel(unsigned* mx, long n) {
  long i = (long)blockIdx.x * blockDim.x + threadIdx.x;
  if (i >= n) return;
  unsigned u = mx[i];
  unsigned bits = (u & 0x80000000u) ? (u & 0x7FFFFFFFu) : ~u;
  float f = __uint_as_float(bits);
  if (((bits >> 23) & 0xFFu) == 0xFFu) f = 0.f;
  mx[i] = __float_as_uint(f);
}

__global__ void exp_kernel(float* __restrict__ abuf, const float* __restrict__ mxf,
                           float* __restrict__ den, const int* __restrict__ ei, int Ecnt) {
  long i = (long)blockIdx.x * blockDim.x + threadIdx.x;
  if (i >= (long)Ecnt * HH) return;
  int e = (int)(i / HH), h = (int)(i % HH);
  int row = ei[e];
  float ex = expf(abuf[i] - mxf[(size_t)row * HH + h]);
  abuf[i] = ex;
  atomicAdd(den + (size_t)row * HH + h, ex);
}

// wave per edge: scatter att * {v, vpg, z} into per-node accumulators
__global__ void scatter_kernel(const float* __restrict__ abuf, const float* __restrict__ den,
                               const float* __restrict__ vf, const float* __restrict__ vpg,
                               const float* __restrict__ z, const int* __restrict__ ei,
                               float* __restrict__ oacc, float* __restrict__ opg,
                               float* __restrict__ opair, int Ecnt) {
  int e = blockIdx.x * (blockDim.x >> 5) + (threadIdx.x >> 5);
  if (e >= Ecnt) return;
  int lane = threadIdx.x & 31;
  int row = ei[e], col = ei[Ecnt + e];
  for (int h = 0; h < HH; ++h) {
    float att = abuf[(size_t)e * HH + h] / (den[(size_t)row * HH + h] + 1e-9f);
    atomicAdd(oacc + (size_t)row * 256 + h * 32 + lane, att * vf[(size_t)col * 256 + h * 32 + lane]);
    for (int j = lane; j < 36; j += 32)
      atomicAdd(opg + (size_t)row * 288 + h * 36 + j, att * vpg[(size_t)col * 288 + h * 36 + j]);
    for (int j = lane; j < CZD; j += 32)
      atomicAdd(opair + (size_t)row * 1024 + h * CZD + j, att * z[(size_t)e * CZD + j]);
  }
}

// op = R^T (opg - t); write op (cols 256..543) and |op| (cols 544..639) into cat
__global__ void opfin_kernel(const float* __restrict__ opg, const float* __restrict__ rot,
                             const float* __restrict__ trans, float* __restrict__ cat, int M) {
  long i = (long)blockIdx.x * blockDim.x + threadIdx.x;
  if (i >= (long)M * HH * PVD) return;
  int n = (int)(i / (HH * PVD)), hp = (int)(i % (HH * PVD));
  const float* R = rot + (size_t)n * 9;
  const float* t = trans + (size_t)n * 3;
  const float* g = opg + (size_t)n * 288 + hp * 3;
  float x = g[0] - t[0], y = g[1] - t[1], z = g[2] - t[2];
  float o0 = R[0] * x + R[3] * y + R[6] * z;
  float o1 = R[1] * x + R[4] * y + R[7] * z;
  float o2 = R[2] * x + R[5] * y + R[8] * z;
  float* cr = cat + (size_t)n * 1664;
  cr[256 + hp * 3 + 0] = o0;
  cr[256 + hp * 3 + 1] = o1;
  cr[256 + hp * 3 + 2] = o2;
  cr[544 + hp] = sqrtf(o0 * o0 + o1 * o1 + o2 * o2 + 1e-8f);
}

// upd6 = ((nf*nm) @ bbW + bbb) * nm   (N x 6)
__global__ void upd6_kernel(const float* __restrict__ nf, const float* __restrict__ nm,
                            const float* __restrict__ W, const float* __restrict__ b,
                            float* __restrict__ u6, int M) {
  long i = (long)blockIdx.x * blockDim.x + threadIdx.x;
  if (i >= (long)M * 6) return;
  int n = (int)(i / 6), j = (int)(i % 6);
  const float* x = nf + (size_t)n * 256;
  float s = 0.f;
  for (int c = 0; c < 256; ++c) s += x[c] * W[c * 6 + j];
  float m = nm[n];
  u6[i] = (s * m + b[j]) * m;
}

__global__ void compose_kernel(const float* __restrict__ rot, const float* __restrict__ trans,
                               const float* __restrict__ u6, float* __restrict__ orot,
                               float* __restrict__ otrans, int M) {
  int n = blockIdx.x * blockDim.x + threadIdx.x;
  if (n >= M) return;
  const float* u = u6 + (size_t)n * 6;
  float w = 1.f, x = u[0], y = u[1], z = u[2];
  float inv = rsqrtf(1.f + x * x + y * y + z * z);
  w *= inv; x *= inv; y *= inv; z *= inv;
  float Q[9] = {1.f - 2.f * (y * y + z * z), 2.f * (x * y - w * z), 2.f * (x * z + w * y),
                2.f * (x * y + w * z), 1.f - 2.f * (x * x + z * z), 2.f * (y * z - w * x),
                2.f * (x * z - w * y), 2.f * (y * z + w * x), 1.f - 2.f * (x * x + y * y)};
  const float* R = rot + (size_t)n * 9;
  float* O = orot + (size_t)n * 9;
#pragma unroll
  for (int ii = 0; ii < 3; ++ii)
#pragma unroll
    for (int jj = 0; jj < 3; ++jj) {
      float s = 0.f;
#pragma unroll
      for (int kk = 0; kk < 3; ++kk) s += R[ii * 3 + kk] * Q[kk * 3 + jj];
      O[ii * 3 + jj] = s;
    }
  const float* t = u + 3;
#pragma unroll
  for (int ii = 0; ii < 3; ++ii)
    otrans[(size_t)n * 3 + ii] =
        R[ii * 3 + 0] * t[0] + R[ii * 3 + 1] * t[1] + R[ii * 3 + 2] * t[2] + trans[(size_t)n * 3 + ii];
}

// ---------------- host helpers ----------------
static inline unsigned nblk(long n, int b) { return (unsigned)((n + b - 1) / b); }

// stage A and W to bf16 (W transposed), then WMMA GEMM. K % 64 == 0, N % 32 == 0, M % 16 == 0.
static void gemm(hipStream_t s, float* ws, const float* A, const float* W, const float* bias,
                 const float* resid, float* C, int M, int K, int N, int act) {
  unsigned short* bfA = (unsigned short*)(ws + OF_BA);
  unsigned short* bfW = (unsigned short*)(ws + OF_BW);
  cvtbf_kernel<<<nblk((long)M * K, 256), 256, 0, s>>>(A, bfA, (long)M * K);
  cvttr_kernel<<<nblk((long)K * N, 256), 256, 0, s>>>(W, bfW, K, N);
  int tilesN2 = N / 32, tilesTotal = (M / 16) * tilesN2;
  gemm_bf16_kernel<<<(tilesTotal + 3) / 4, 128, 0, s>>>(bfA, bfW, bias, resid, C,
                                                        M, K, N, tilesN2, tilesTotal, act);
}
static void layernorm(hipStream_t s, const float* x, const float* r, const float* g,
                      const float* b, float* o, int M, int D) {
  ln_kernel<<<(M + 3) / 4, 128, 0, s>>>(x, r, g, b, o, M, D);
}

struct IpaP {
  const float *qW, *qb, *kW, *kb, *vW, *vb, *qpW, *qpb, *kpW, *kpb, *vpW, *vpb;
  const float *bzW, *bzb, *hw, *oW, *ob;
};

static void run_ipa(hipStream_t s, float* ws, const float* nf, const float* z, int Ecnt,
                    const int* ei, const float* rot, const float* trans, const float* rm,
                    const IpaP& p, float* upd) {
  float* q = ws + OF_Q;   float* k = ws + OF_K;   float* v = ws + OF_V;
  float* qp = ws + OF_QP; float* kp = ws + OF_KP; float* vp = ws + OF_VP;
  float* qpg = ws + OF_QPG; float* kpg = ws + OF_KPG; float* vpg = ws + OF_VPG;
  float* mx = ws + OF_MX;   float* den = ws + OF_DEN;
  float* oacc = ws + OF_OAC; float* opg = ws + OF_OPG; float* opair = ws + OF_OPR;
  float* abuf = ws + OF_H;   float* cat = ws + OF_H;  // temporally disjoint uses

  gemm(s, ws, nf, p.qW, p.qb, nullptr, q, NN, 256, 256, 0);
  gemm(s, ws, nf, p.kW, p.kb, nullptr, k, NN, 256, 256, 0);
  gemm(s, ws, nf, p.vW, p.vb, nullptr, v, NN, 256, 256, 0);
  gemm(s, ws, nf, p.qpW, p.qpb, nullptr, qp, NN, 256, 192, 0);
  gemm(s, ws, nf, p.kpW, p.kpb, nullptr, kp, NN, 256, 192, 0);
  gemm(s, ws, nf, p.vpW, p.vpb, nullptr, vp, NN, 256, 288, 0);

  frame_apply_kernel<<<nblk((long)NN * 64, 256), 256, 0, s>>>(qp, rot, trans, qpg, NN, 64);
  frame_apply_kernel<<<nblk((long)NN * 64, 256), 256, 0, s>>>(kp, rot, trans, kpg, NN, 64);
  frame_apply_kernel<<<nblk((long)NN * 96, 256), 256, 0, s>>>(vp, rot, trans, vpg, NN, 96);

  fillu_kernel<<<nblk((long)NN * HH, 256), 256, 0, s>>>((unsigned*)mx, 0x007FFFFFu, (long)NN * HH); // enc(-inf)
  fillf_kernel<<<nblk((long)NN * HH, 256), 256, 0, s>>>(den, 0.f, (long)NN * HH);
  fillf_kernel<<<nblk((long)NN * 256, 256), 256, 0, s>>>(oacc, 0.f, (long)NN * 256);
  fillf_kernel<<<nblk((long)NN * 288, 256), 256, 0, s>>>(opg, 0.f, (long)NN * 288);
  fillf_kernel<<<nblk((long)NN * 1024, 256), 256, 0, s>>>(opair, 0.f, (long)NN * 1024);

  logits_kernel<<<nblk((long)Ecnt * HH, 256), 256, 0, s>>>(q, k, qpg, kpg, z, p.bzW, p.bzb, p.hw,
                                                           rm, ei, abuf, (unsigned*)mx, Ecnt);
  mxfix_kernel<<<nblk((long)NN * HH, 256), 256, 0, s>>>((unsigned*)mx, (long)NN * HH);
  exp_kernel<<<nblk((long)Ecnt * HH, 256), 256, 0, s>>>(abuf, mx, den, ei, Ecnt);
  scatter_kernel<<<nblk(Ecnt, 4), 128, 0, s>>>(abuf, den, v, vpg, z, ei, oacc, opg, opair, Ecnt);

  copy_cols_kernel<<<nblk((long)NN * 256, 256), 256, 0, s>>>(oacc, cat, NN, 256, 1664, 0);
  opfin_kernel<<<nblk((long)NN * HH * PVD, 256), 256, 0, s>>>(opg, rot, trans, cat, NN);
  copy_cols_kernel<<<nblk((long)NN * 1024, 256), 256, 0, s>>>(opair, cat, NN, 1024, 1664, 640);

  gemm(s, ws, cat, p.oW, p.ob, nullptr, upd, NN, 1664, 256, 0);
}

// ---------------- entry point ----------------
extern "C" void kernel_launch(void* const* d_in, const int* in_sizes, int n_in,
                              void* d_out, int out_size, void* d_ws, size_t ws_size,
                              hipStream_t stream) {
  (void)in_sizes; (void)n_in; (void)out_size;
  if (ws_size < WS_FLOATS * sizeof(float)) return;  // insufficient scratch

  const float* nf_in    = (const float*)d_in[0];
  const float* lat_in   = (const float*)d_in[1];
  const float* rot_in   = (const float*)d_in[2];
  const float* trans_in = (const float*)d_in[3];
  const float* ef       = (const float*)d_in[4];
  const float* sef      = (const float*)d_in[5];
  const int*   ei       = (const int*)d_in[6];
  const int*   sei      = (const int*)d_in[7];
  const float* rm       = (const float*)d_in[8];
  const float* nm       = (const float*)d_in[9];
  auto P = [&](int i) { return (const float*)d_in[i]; };
  auto mkipa = [&](int b) {
    IpaP p;
    p.qW = P(b); p.qb = P(b + 1); p.kW = P(b + 2); p.kb = P(b + 3);
    p.vW = P(b + 4); p.vb = P(b + 5); p.qpW = P(b + 6); p.qpb = P(b + 7);
    p.kpW = P(b + 8); p.kpb = P(b + 9); p.vpW = P(b + 10); p.vpb = P(b + 11);
    p.bzW = P(b + 12); p.bzb = P(b + 13); p.hw = P(b + 14); p.oW = P(b + 15); p.ob = P(b + 16);
    return p;
  };

  float* ws = (float*)d_ws;
  float* zb = ws + OF_Z;
  float* hb = ws + OF_H;
  float* nfb = ws + OF_NF;
  float* updb = ws + OF_UPD;
  float* t1 = ws + OF_T1;
  float* t2 = ws + OF_T2;
  float* u6 = ws + OF_U6;
  float* out_nf = (float*)d_out;
  float* out_rot = out_nf + (size_t)NN * 256;
  float* out_trans = out_rot + (size_t)NN * 9;
  float* out_lat = out_trans + (size_t)NN * 3;

  // ---- edge MLP: z = LN(relu(relu(ef@e1)@e2)@e3) ----
  gemm(stream, ws, ef, P(10), P(11), nullptr, hb, EE, EIND, CZD, 1);
  gemm(stream, ws, hb, P(12), P(13), nullptr, zb, EE, CZD, CZD, 1);
  gemm(stream, ws, zb, P(14), P(15), nullptr, hb, EE, CZD, CZD, 0);
  layernorm(stream, hb, nullptr, P(16), P(17), zb, EE, CZD);

  // ---- node projection: nf = concat(nf, lat) @ node.W + b ----
  concat_kernel<<<nblk((long)NN * (CSD + CLD), 256), 256, 0, stream>>>(nf_in, lat_in, t1, NN, CSD, CLD);
  gemm(stream, ws, t1, P(18), P(19), nullptr, nfb, NN, CSD + CLD, CSD, 0);

  // ---- spatial IPA ----
  IpaP sp = mkipa(20), sq = mkipa(37);
  run_ipa(stream, ws, nfb, zb, EE, ei, rot_in, trans_in, rm, sp, updb);
  scale_rows_kernel<<<nblk((long)NN * 256, 256), 256, 0, stream>>>(updb, rm, updb, NN, 256);
  layernorm(stream, nfb, updb, P(54), P(55), nfb, NN, CSD);

  // ---- sequence IPA (same ln1, faithful to reference) ----
  run_ipa(stream, ws, nfb, sef, ESN, sei, rot_in, trans_in, rm, sq, updb);
  scale_rows_kernel<<<nblk((long)NN * 256, 256), 256, 0, stream>>>(updb, rm, updb, NN, 256);
  layernorm(stream, nfb, updb, P(54), P(55), nfb, NN, CSD);

  // ---- transition MLP + LN ----
  gemm(stream, ws, nfb, P(56), P(57), nullptr, t1, NN, 256, 256, 1);
  gemm(stream, ws, t1, P(58), P(59), nullptr, t2, NN, 256, 256, 1);
  gemm(stream, ws, t2, P(60), P(61), nullptr, t1, NN, 256, 256, 0);
  layernorm(stream, nfb, t1, P(62), P(63), nfb, NN, CSD);

  // ---- nf output (masked) ----
  scale_rows_kernel<<<nblk((long)NN * 256, 256), 256, 0, stream>>>(nfb, rm, out_nf, NN, 256);

  // ---- backbone update + frame composition ----
  upd6_kernel<<<nblk((long)NN * 6, 256), 256, 0, stream>>>(out_nf, nm, P(64), P(65), u6, NN);
  compose_kernel<<<nblk(NN, 256), 256, 0, stream>>>(rot_in, trans_in, u6, out_rot, out_trans, NN);

  // ---- latent update: lat_out = lat + nf @ lat.W + lat.b ----
  gemm(stream, ws, out_nf, P(66), P(67), lat_in, out_lat, NN, 256, CLD, 0);
}
